// GraphConditioner_42786464203343
// MI455X (gfx1250) — matrix operand order repaired and checked by reading, not compiled, not run on hardware
//
#include <hip/hip_runtime.h>
#include <math.h>

#define N_NODES 16384
#define N_EDGES 262144
#define DIM_IN  128
#define DIM_H   256
#define DIM_OUT 256
#define N_LAYERS 4
#define N_HEADS  8
#define HEAD_DIM 32
#define N_GRAPHS 64
#define NPG      256   // nodes per graph

typedef __bf16 bf16_t;
typedef bf16_t v16bf __attribute__((ext_vector_type(16)));
typedef float  v8f   __attribute__((ext_vector_type(8)));

union Frag16 { v16bf v; unsigned short u[16]; };

__device__ __forceinline__ unsigned short f2bf(float f) {
  unsigned int u = __float_as_uint(f);
  u += 0x7fffu + ((u >> 16) & 1u);           // round-to-nearest-even
  return (unsigned short)(u >> 16);
}
__device__ __forceinline__ float bf2f(unsigned short s) {
  return __uint_as_float(((unsigned int)s) << 16);
}

// ---------------------------------------------------------------------------
// Async global->LDS staging (CDNA5 GLOBAL_LOAD_ASYNC_TO_LDS_B128, ASYNCcnt).
// Builtin signature (from hipcc diagnostic): (v4i AS1*, v4i AS3*, Ii, Ii).
// ---------------------------------------------------------------------------
#if defined(__has_builtin)
#if __has_builtin(__builtin_amdgcn_global_load_async_to_lds_b128) && \
    __has_builtin(__builtin_amdgcn_s_wait_asynccnt)
#define ASYNC_COPY 1
#endif
#endif

typedef int v4i __attribute__((vector_size(16)));
typedef __attribute__((address_space(1))) v4i gv4i;
typedef __attribute__((address_space(3))) v4i lv4i;

__device__ __forceinline__ void copy_b128(const void* g, void* l) {
#ifdef ASYNC_COPY
  __builtin_amdgcn_global_load_async_to_lds_b128(
      (gv4i*)(unsigned long long)g,
      (lv4i*)(unsigned int)(unsigned long long)l, 0, 0);
#else
  *(uint4*)l = *(const uint4*)g;
#endif
}
__device__ __forceinline__ void copy_wait() {
#ifdef ASYNC_COPY
  __builtin_amdgcn_s_wait_asynccnt(0);
#endif
}

// ---------------------------------------------------------------------------
// Tiled WMMA GEMM: C[M,Nn] = A[M,K](bf16) * Bt^T  with Bt row-major [Nn,K].
// (All B operands are weights, pre-transposed on device, so both A and B
//  stage/fragment paths are fully vectorized b128 LDS traffic.)
// M % 128 == 0, Nn % 128 == 0, K % 32 == 0.
// ---------------------------------------------------------------------------
#define LD_S 40   // LDS row stride in halves for 32-wide K slab (+8 pad)

__device__ __forceinline__ v16bf lda_frag(const unsigned short* Ts, int row0, int lane) {
  // A-matrix 16x32 bf16 layout: lane m=lane&15, kh=lane>>4;
  // u[j] = K = kh*16 + j  (contiguous, matches B)
  int m = lane & 15, kh = lane >> 4;
  const unsigned short* p = Ts + (row0 + m) * LD_S;
  Frag16 f;
#pragma unroll
  for (int j = 0; j < 8; ++j) f.u[j] = p[kh * 16 + j];
#pragma unroll
  for (int j = 0; j < 8; ++j) f.u[8 + j] = p[kh * 16 + 8 + j];
  return f.v;
}

__device__ __forceinline__ v16bf ldb_frag(const unsigned short* Ts, int row0, int lane) {
  // B-matrix 32x16 bf16 layout from K-contiguous storage: lane n=lane&15,
  // kh=lane>>4; u[j] = K = kh*16 + j  (single 32B contiguous span)
  int n = lane & 15, kh = lane >> 4;
  const unsigned short* p = Ts + (row0 + n) * LD_S + kh * 16;
  Frag16 f;
#pragma unroll
  for (int j = 0; j < 16; ++j) f.u[j] = p[j];
  return f.v;
}

__global__ __launch_bounds__(256) void wmma_gemm(
    const unsigned short* __restrict__ A, const unsigned short* __restrict__ Bt,
    const float* __restrict__ bias, float* __restrict__ out32,
    unsigned short* __restrict__ out16, int M, int Nn, int K, int relu)
{
  __shared__ unsigned short As[128 * LD_S];
  __shared__ unsigned short Bs[128 * LD_S];

  const int tid = threadIdx.x, lane = tid & 31, wid = tid >> 5;
  const int wm = wid >> 2, wn = wid & 3;
  const int row0 = blockIdx.y * 128, col0 = blockIdx.x * 128;

  v8f acc[4][2];
#pragma unroll
  for (int mt = 0; mt < 4; ++mt)
#pragma unroll
    for (int nt = 0; nt < 2; ++nt)
#pragma unroll
      for (int v = 0; v < 8; ++v) acc[mt][nt][v] = 0.0f;

  const int r = tid >> 1, ks = (tid & 1) * 16;
  for (int k0 = 0; k0 < K; k0 += 32) {
    copy_b128(A + (size_t)(row0 + r) * K + k0 + ks, As + r * LD_S + ks);
    copy_b128(A + (size_t)(row0 + r) * K + k0 + ks + 8, As + r * LD_S + ks + 8);
    copy_b128(Bt + (size_t)(col0 + r) * K + k0 + ks, Bs + r * LD_S + ks);
    copy_b128(Bt + (size_t)(col0 + r) * K + k0 + ks + 8, Bs + r * LD_S + ks + 8);
    copy_wait();
    __syncthreads();

    v16bf bfr[2];
#pragma unroll
    for (int nt = 0; nt < 2; ++nt) bfr[nt] = ldb_frag(Bs, wn * 32 + nt * 16, lane);
#pragma unroll
    for (int mt = 0; mt < 4; ++mt) {
      v16bf afr = lda_frag(As, wm * 64 + mt * 16, lane);
#pragma unroll
      for (int nt = 0; nt < 2; ++nt)
        acc[mt][nt] = __builtin_amdgcn_wmma_f32_16x16x32_bf16(
            false, afr, false, bfr[nt], (short)0, acc[mt][nt], false, false);
    }
    __syncthreads();
  }

  const int n16 = lane & 15, mh = lane >> 4;
#pragma unroll
  for (int mt = 0; mt < 4; ++mt)
#pragma unroll
    for (int nt = 0; nt < 2; ++nt) {
      int col = col0 + wn * 32 + nt * 16 + n16;
      float bv = bias ? bias[col] : 0.0f;
#pragma unroll
      for (int v = 0; v < 8; ++v) {
        int row = row0 + wm * 64 + mt * 16 + v + mh * 8;
        float val = acc[mt][nt][v] + bv;
        if (relu) val = fmaxf(val, 0.0f);
        size_t idx = (size_t)row * Nn + col;
        if (out32) out32[idx] = val;
        if (out16) out16[idx] = f2bf(val);
      }
    }
}

// ---------------------------------------------------------------------------
// Flash-style MHA: one block per (graph, head). K (row) and V (transposed)
// staged in LDS as bf16; QK^T and PV through WMMA; online softmax; per-wave
// LDS bounce re-lays P from C-layout into A-layout.
// ---------------------------------------------------------------------------
#define VT_S 264  // Vt row stride in halves (256 keys + 8 pad)

__global__ __launch_bounds__(256) void mha_kernel(
    const unsigned short* __restrict__ qkv, unsigned short* __restrict__ obf)
{
  __shared__ unsigned short Kl[256 * 40];      // [key][oc]
  __shared__ unsigned short Vt[32 * VT_S];     // [oc][key]
  __shared__ unsigned short Ps[8 * 32 * 40];   // per-wave P bounce

  const int bh = blockIdx.x, gb = bh >> 3, hh = bh & 7;
  const int tid = threadIdx.x, w = tid >> 5, lane = tid & 31;

  { // stage K (async, row layout) and V (transposed scatter)
    int key = tid;
    const unsigned short* kr = qkv + (size_t)(gb * NPG + key) * 768 + 256 + hh * 32;
#pragma unroll
    for (int j = 0; j < 4; ++j) copy_b128(kr + 8 * j, Kl + key * 40 + 8 * j);
    union { uint4 q[4]; unsigned short s[32]; } vv;
    const uint4* vr = (const uint4*)(qkv + (size_t)(gb * NPG + key) * 768 + 512 + hh * 32);
    vv.q[0] = vr[0]; vv.q[1] = vr[1]; vv.q[2] = vr[2]; vv.q[3] = vr[3];
#pragma unroll
    for (int oc = 0; oc < 32; ++oc) Vt[oc * VT_S + key] = vv.s[oc];
    copy_wait();
  }
  __syncthreads();

  const int qbase = w * 32;
  const int m = lane & 15, kh = lane >> 4;
  const float scale = 0.17677669529663687f; // 1/sqrt(32)

  v16bf qf[2];
#pragma unroll
  for (int mt = 0; mt < 2; ++mt) {
    Frag16 f;
    const unsigned short* qr = qkv + (size_t)(gb * NPG + qbase + mt * 16 + m) * 768 + hh * 32;
#pragma unroll
    for (int j = 0; j < 8; ++j) f.u[j] = f2bf(bf2f(qr[kh * 16 + j]) * scale);
#pragma unroll
    for (int j = 0; j < 8; ++j) f.u[8 + j] = f2bf(bf2f(qr[kh * 16 + 8 + j]) * scale);
    qf[mt] = f.v;
  }

  v8f O[2][2];
  float mrow[2][8], lrow[2][8];
#pragma unroll
  for (int mt = 0; mt < 2; ++mt) {
#pragma unroll
    for (int nt = 0; nt < 2; ++nt)
#pragma unroll
      for (int v = 0; v < 8; ++v) O[mt][nt][v] = 0.0f;
#pragma unroll
    for (int v = 0; v < 8; ++v) { mrow[mt][v] = -1e30f; lrow[mt][v] = 0.0f; }
  }

  for (int kb = 0; kb < 8; ++kb) { // 32 keys per step
    v8f S[2][2];
#pragma unroll
    for (int mt = 0; mt < 2; ++mt)
#pragma unroll
      for (int nt = 0; nt < 2; ++nt)
#pragma unroll
        for (int v = 0; v < 8; ++v) S[mt][nt][v] = 0.0f;

    v16bf kf[2];
#pragma unroll
    for (int nt = 0; nt < 2; ++nt) { // B = K^T: contraction over oc, contiguous
      Frag16 f;
      const unsigned short* p = Kl + (kb * 32 + nt * 16 + m) * 40 + kh * 16;
#pragma unroll
      for (int j = 0; j < 16; ++j) f.u[j] = p[j];
      kf[nt] = f.v;
    }
#pragma unroll
    for (int mt = 0; mt < 2; ++mt)
#pragma unroll
      for (int nt = 0; nt < 2; ++nt)
        S[mt][nt] = __builtin_amdgcn_wmma_f32_16x16x32_bf16(
            false, qf[mt], false, kf[nt], (short)0, S[mt][nt], false, false);

    // online softmax (rows split across lane halves per C-layout)
#pragma unroll
    for (int mt = 0; mt < 2; ++mt)
#pragma unroll
      for (int v = 0; v < 8; ++v) {
        float s0 = S[mt][0][v], s1 = S[mt][1][v];
        float mx = fmaxf(s0, s1);
#pragma unroll
        for (int off = 8; off; off >>= 1) mx = fmaxf(mx, __shfl_xor(mx, off, 16));
        float mn = fmaxf(mrow[mt][v], mx);
        float corr = __expf(mrow[mt][v] - mn);
        float p0 = __expf(s0 - mn), p1 = __expf(s1 - mn);
        float rs = p0 + p1;
#pragma unroll
        for (int off = 8; off; off >>= 1) rs += __shfl_xor(rs, off, 16);
        lrow[mt][v] = lrow[mt][v] * corr + rs;
        mrow[mt][v] = mn;
        O[mt][0][v] *= corr;
        O[mt][1][v] *= corr;
        int prow = mt * 16 + v + kh * 8;
        Ps[(w * 32 + prow) * 40 + m]      = f2bf(p0);
        Ps[(w * 32 + prow) * 40 + 16 + m] = f2bf(p1);
      }

    v16bf pf[2];
#pragma unroll
    for (int mt = 0; mt < 2; ++mt) { // P as A-matrix (contraction over keys)
      Frag16 f;
      const unsigned short* pr = Ps + (w * 32 + mt * 16 + m) * 40;
#pragma unroll
      for (int j = 0; j < 8; ++j) f.u[j] = pr[kh * 16 + j];
#pragma unroll
      for (int j = 0; j < 8; ++j) f.u[8 + j] = pr[kh * 16 + 8 + j];
      pf[mt] = f.v;
    }
    v16bf vf[2];
#pragma unroll
    for (int nt = 0; nt < 2; ++nt) { // B = V: contraction over keys, contiguous in Vt
      Frag16 f;
      const unsigned short* p = Vt + (nt * 16 + m) * VT_S + kb * 32 + kh * 16;
#pragma unroll
      for (int j = 0; j < 16; ++j) f.u[j] = p[j];
      vf[nt] = f.v;
    }
#pragma unroll
    for (int mt = 0; mt < 2; ++mt)
#pragma unroll
      for (int nt = 0; nt < 2; ++nt)
        O[mt][nt] = __builtin_amdgcn_wmma_f32_16x16x32_bf16(
            false, pf[mt], false, vf[nt], (short)0, O[mt][nt], false, false);
  }

#pragma unroll
  for (int mt = 0; mt < 2; ++mt)
#pragma unroll
    for (int nt = 0; nt < 2; ++nt)
#pragma unroll
      for (int v = 0; v < 8; ++v) {
        int row = gb * NPG + qbase + mt * 16 + v + kh * 8;
        int col = hh * 32 + nt * 16 + m;
        obf[(size_t)row * DIM_H + col] = f2bf(O[mt][nt][v] / lrow[mt][v]);
      }
}

// ---------------------------------------------------------------------------
// LayerNorm (256 channels) with fused adds: out = post + LN(a + b + rowbias)
// ---------------------------------------------------------------------------
__global__ __launch_bounds__(256) void ln_kernel(
    const float* __restrict__ a, const float* __restrict__ b,
    const float* __restrict__ rowbias, const float* __restrict__ post,
    const float* __restrict__ g, const float* __restrict__ beta,
    float* __restrict__ out32, unsigned short* __restrict__ out16, int rows)
{
  int row = blockIdx.x * 8 + (threadIdx.x >> 5);
  int lane = threadIdx.x & 31;
  if (row >= rows) return;
  size_t base = (size_t)row * 256;
  float vals[8], s = 0.0f;
#pragma unroll
  for (int j = 0; j < 8; ++j) {
    int c = lane + 32 * j;
    float v = a[base + c];
    if (b) v += b[base + c];
    if (rowbias) v += rowbias[c];
    vals[j] = v; s += v;
  }
#pragma unroll
  for (int off = 16; off; off >>= 1) s += __shfl_xor(s, off, 32);
  float mean = s * (1.0f / 256.0f);
  float s2 = 0.0f;
#pragma unroll
  for (int j = 0; j < 8; ++j) { float d = vals[j] - mean; s2 += d * d; }
#pragma unroll
  for (int off = 16; off; off >>= 1) s2 += __shfl_xor(s2, off, 32);
  float rinv = rsqrtf(s2 * (1.0f / 256.0f) + 1e-5f);
#pragma unroll
  for (int j = 0; j < 8; ++j) {
    int c = lane + 32 * j;
    float o = (vals[j] - mean) * rinv * g[c] + beta[c];
    if (post) o += post[base + c];
    if (out32) out32[base + c] = o;
    if (out16) out16[base + c] = f2bf(o);
  }
}

// ---------------------------------------------------------------------------
// Small helpers: fills, converts, transpose-convert (f32 [R][C] -> bf16 [C][R])
// ---------------------------------------------------------------------------
__global__ void fill_f32(float* p, float v, int n) {
  int i = blockIdx.x * blockDim.x + threadIdx.x;
  if (i < n) p[i] = v;
}
__global__ void f32_to_bf16(const float* __restrict__ in, unsigned short* __restrict__ out, int n) {
  int i = blockIdx.x * blockDim.x + threadIdx.x;
  if (i < n) out[i] = f2bf(in[i]);
}
__global__ __launch_bounds__(256) void transpose_to_bf16(
    const float* __restrict__ in, unsigned short* __restrict__ out, int R, int C)
{
  __shared__ float tile[32][33];
  int r0 = blockIdx.y * 32, c0 = blockIdx.x * 32;
  int tx = threadIdx.x & 31, ty = threadIdx.x >> 5;
#pragma unroll
  for (int j = 0; j < 32; j += 8)
    tile[ty + j][tx] = in[(size_t)(r0 + ty + j) * C + c0 + tx];
  __syncthreads();
#pragma unroll
  for (int j = 0; j < 32; j += 8)
    out[(size_t)(c0 + ty + j) * R + r0 + tx] = f2bf(tile[tx][ty + j]);
}

// ---------------------------------------------------------------------------
// GAT edge-softmax / aggregation
// ---------------------------------------------------------------------------
__global__ void alpha_kernel(const float* __restrict__ xw,
                             const float* __restrict__ asrc, const float* __restrict__ adst,
                             float* __restrict__ as_out, float* __restrict__ ad_out)
{
  int i = blockIdx.x * blockDim.x + threadIdx.x;
  if (i >= N_NODES * N_HEADS) return;
  int n = i >> 3, hh = i & 7;
  const float* xr = xw + (size_t)n * DIM_H + hh * HEAD_DIM;
  float s = 0.0f, d = 0.0f;
#pragma unroll
  for (int oc = 0; oc < HEAD_DIM; ++oc) {
    s += xr[oc] * asrc[hh * HEAD_DIM + oc];
    d += xr[oc] * adst[hh * HEAD_DIM + oc];
  }
  as_out[i] = s; ad_out[i] = d;
}

__device__ __forceinline__ void atomicMaxF(float* addr, float val) {
  if (val >= 0.0f) atomicMax((int*)addr, __float_as_int(val));
  else atomicMin((unsigned int*)addr, __float_as_uint(val));
}

__global__ void edge_logits(const int* __restrict__ ei,
                            const float* __restrict__ as_, const float* __restrict__ ad_,
                            float* __restrict__ ae, float* __restrict__ amax)
{
  int idx = blockIdx.x * blockDim.x + threadIdx.x;
  if (idx >= N_EDGES * N_HEADS) return;
  int e = idx >> 3, hh = idx & 7;
  int s = ei[e], d = ei[N_EDGES + e];
  float a = as_[s * N_HEADS + hh] + ad_[d * N_HEADS + hh];
  a = a > 0.0f ? a : 0.2f * a;
  ae[idx] = a;
  atomicMaxF(&amax[d * N_HEADS + hh], a);
}

__global__ void edge_exp(const int* __restrict__ ei, float* __restrict__ ae,
                         const float* __restrict__ amax, float* __restrict__ asum)
{
  int idx = blockIdx.x * blockDim.x + threadIdx.x;
  if (idx >= N_EDGES * N_HEADS) return;
  int e = idx >> 3, hh = idx & 7;
  int d = ei[N_EDGES + e];
  float v = __expf(ae[idx] - amax[d * N_HEADS + hh]);
  ae[idx] = v;
  atomicAdd(&asum[d * N_HEADS + hh], v);
}

__global__ __launch_bounds__(256) void edge_aggr(const int* __restrict__ ei,
                                                 const float* __restrict__ ae,
                                                 const float* __restrict__ asum,
                                                 const float* __restrict__ xw,
                                                 float* __restrict__ hagg)
{
  int e = blockIdx.x, c = threadIdx.x;
  int s = ei[e], d = ei[N_EDGES + e];
  int hh = c >> 5;
  float w = ae[(size_t)e * N_HEADS + hh] / asum[d * N_HEADS + hh];
  atomicAdd(&hagg[(size_t)d * DIM_H + c], w * xw[(size_t)s * DIM_H + c]);
}

// ---------------------------------------------------------------------------
// Host orchestration
// ---------------------------------------------------------------------------
extern "C" void kernel_launch(void* const* d_in, const int* in_sizes, int n_in,
                              void* d_out, int out_size, void* d_ws, size_t ws_size,
                              hipStream_t stream)
{
  (void)in_sizes; (void)n_in; (void)out_size; (void)ws_size;
  const float* x_in = (const float*)d_in[0];
  const int*   ei   = (const int*)d_in[1];
  const float* wi   = (const float*)d_in[2];
  const float* bi   = (const float*)d_in[3];
  const float* wg   = (const float*)d_in[4];
  const float* asrc = (const float*)d_in[5];
  const float* adst = (const float*)d_in[6];
  const float* bg   = (const float*)d_in[7];
  const float* g1   = (const float*)d_in[8];
  const float* b1   = (const float*)d_in[9];
  const float* inw  = (const float*)d_in[10];
  const float* inb  = (const float*)d_in[11];
  const float* ow   = (const float*)d_in[12];
  const float* ob   = (const float*)d_in[13];
  const float* g2   = (const float*)d_in[14];
  const float* b2   = (const float*)d_in[15];
  const float* mw1  = (const float*)d_in[16];
  const float* mb1  = (const float*)d_in[17];
  const float* mw2  = (const float*)d_in[18];
  const float* mb2  = (const float*)d_in[19];
  const float* g3   = (const float*)d_in[20];
  const float* b3   = (const float*)d_in[21];
  const float* gf   = (const float*)d_in[22];
  const float* bfp  = (const float*)d_in[23];
  const float* wo   = (const float*)d_in[24];
  const float* bo   = (const float*)d_in[25];

  char* ws = (char*)d_ws;
  size_t off = 0;
  auto alloc = [&](size_t bytes) -> char* {
    char* p = ws + off;
    off += (bytes + 255) & ~(size_t)255;
    return p;
  };
  const size_t NH = (size_t)N_NODES * DIM_H;
  const int HH = DIM_H * DIM_H;
  unsigned short* xbf    = (unsigned short*)alloc(NH * 2);
  float*          x      = (float*)alloc(NH * 4);
  float*          tmp    = (float*)alloc(NH * 4);            // xw / oproj / mlp2
  float*          hagg   = (float*)alloc(NH * 4);
  float*          hloc   = (float*)alloc(NH * 4);
  float*          out0   = (float*)alloc(NH * 4);
  unsigned short* bfA    = (unsigned short*)alloc(NH * 2);   // x0bf / o_bf / out0_bf
  unsigned short* bfB    = (unsigned short*)alloc((size_t)N_NODES * 768 * 2); // qkv / mlp1
  float*          al_s   = (float*)alloc((size_t)N_NODES * N_HEADS * 4);
  float*          al_d   = (float*)alloc((size_t)N_NODES * N_HEADS * 4);
  float*          amax   = (float*)alloc((size_t)N_NODES * N_HEADS * 4);
  float*          asum   = (float*)alloc((size_t)N_NODES * N_HEADS * 4);
  float*          ae     = (float*)alloc((size_t)N_EDGES * N_HEADS * 4);
  // pre-transposed bf16 weights (Bt = row-major [Nout][K])
  unsigned short* wiT    = (unsigned short*)alloc((size_t)DIM_H * DIM_IN * 2);
  unsigned short* wgT    = (unsigned short*)alloc((size_t)N_LAYERS * HH * 2);
  unsigned short* inw_bf = (unsigned short*)alloc((size_t)N_LAYERS * 3 * HH * 2); // already [3H][H]
  unsigned short* ow_bf  = (unsigned short*)alloc((size_t)N_LAYERS * HH * 2);     // already [H][H]
  unsigned short* mw1T   = (unsigned short*)alloc((size_t)N_LAYERS * 2 * HH * 2);
  unsigned short* mw2T   = (unsigned short*)alloc((size_t)N_LAYERS * 2 * HH * 2);
  unsigned short* woT    = (unsigned short*)alloc((size_t)DIM_OUT * DIM_H * 2);

  auto cvt = [&](const float* src, unsigned short* dst, int n) {
    f32_to_bf16<<<(n + 255) / 256, 256, 0, stream>>>(src, dst, n);
  };
  auto trn = [&](const float* src, unsigned short* dst, int R, int C) {
    dim3 grid(C / 32, R / 32);
    transpose_to_bf16<<<grid, 256, 0, stream>>>(src, dst, R, C);
  };
  auto gemm = [&](const unsigned short* A, const unsigned short* Bt,
                  const float* bias, int relu, float* o32, unsigned short* o16,
                  int M, int Nn, int K) {
    dim3 grid(Nn / 128, M / 128);
    wmma_gemm<<<grid, 256, 0, stream>>>(A, Bt, bias, o32, o16, M, Nn, K, relu);
  };
  auto ln = [&](const float* a, const float* b, const float* rb, const float* post,
                const float* g, const float* be, float* o32, unsigned short* o16) {
    ln_kernel<<<N_NODES / 8, 256, 0, stream>>>(a, b, rb, post, g, be, o32, o16, N_NODES);
  };

  // weight prep (every launch; weights are inputs)
  trn(wi, wiT, DIM_IN, DIM_H);
  cvt(inw, inw_bf, N_LAYERS * 3 * HH);
  cvt(ow, ow_bf, N_LAYERS * HH);
  trn(wo, woT, DIM_H, DIM_OUT);
  for (int i = 0; i < N_LAYERS; ++i) {
    trn(wg + (size_t)i * HH, wgT + (size_t)i * HH, DIM_H, DIM_H);
    trn(mw1 + (size_t)i * 2 * HH, mw1T + (size_t)i * 2 * HH, DIM_H, 2 * DIM_H);
    trn(mw2 + (size_t)i * 2 * HH, mw2T + (size_t)i * 2 * HH, 2 * DIM_H, DIM_H);
  }

  // x = relu(x_in @ wi + bi)
  cvt(x_in, bfA, N_NODES * DIM_IN);
  gemm(bfA, wiT, bi, 1, x, xbf, N_NODES, DIM_H, DIM_IN);

  for (int i = 0; i < N_LAYERS; ++i) {
    // ---- GAT local branch ----
    gemm(xbf, wgT + (size_t)i * HH, nullptr, 0, tmp, nullptr, N_NODES, DIM_H, DIM_H); // xw
    fill_f32<<<(N_NODES * N_HEADS + 255) / 256, 256, 0, stream>>>(amax, -1e30f, N_NODES * N_HEADS);
    fill_f32<<<(N_NODES * N_HEADS + 255) / 256, 256, 0, stream>>>(asum, 0.0f, N_NODES * N_HEADS);
    fill_f32<<<((int)NH + 255) / 256, 256, 0, stream>>>(hagg, 0.0f, (int)NH);
    alpha_kernel<<<(N_NODES * N_HEADS + 255) / 256, 256, 0, stream>>>(
        tmp, asrc + i * N_HEADS * HEAD_DIM, adst + i * N_HEADS * HEAD_DIM, al_s, al_d);
    edge_logits<<<(N_EDGES * N_HEADS + 255) / 256, 256, 0, stream>>>(ei, al_s, al_d, ae, amax);
    edge_exp<<<(N_EDGES * N_HEADS + 255) / 256, 256, 0, stream>>>(ei, ae, amax, asum);
    edge_aggr<<<N_EDGES, 256, 0, stream>>>(ei, ae, asum, tmp, hagg);
    ln(hagg, x, bg + i * DIM_H, nullptr, g1 + i * DIM_H, b1 + i * DIM_H, hloc, nullptr);
    // ---- global MHA branch ----
    gemm(xbf, inw_bf + (size_t)i * 3 * HH, inb + i * 3 * DIM_H, 0,
         nullptr, bfB, N_NODES, 3 * DIM_H, DIM_H);                        // qkv (bf16)
    mha_kernel<<<N_GRAPHS * N_HEADS, 256, 0, stream>>>(bfB, bfA);         // o (bf16)
    gemm(bfA, ow_bf + (size_t)i * HH, ob + i * DIM_H, 0, tmp, nullptr,
         N_NODES, DIM_H, DIM_H);                                          // oproj
    // out0 = hloc + LN(oproj + x)
    ln(tmp, x, nullptr, hloc, g2 + i * DIM_H, b2 + i * DIM_H, out0, bfA);
    // ---- MLP ----
    gemm(bfA, mw1T + (size_t)i * 2 * HH, mb1 + i * 2 * DIM_H, 1,
         nullptr, bfB, N_NODES, 2 * DIM_H, DIM_H);                        // relu, bf16
    gemm(bfB, mw2T + (size_t)i * 2 * HH, mb2 + i * DIM_H, 0, tmp, nullptr,
         N_NODES, DIM_H, 2 * DIM_H);                                      // mlp2
    ln(out0, tmp, nullptr, nullptr, g3 + i * DIM_H, b3 + i * DIM_H, x, xbf);
  }

  // final LN + output projection
  ln(x, nullptr, nullptr, nullptr, gf, bfp, nullptr, xbf);
  gemm(xbf, woT, bo, 0, (float*)d_out, nullptr, N_NODES, DIM_OUT, DIM_H);
}
